// GNN_45183055954600
// MI455X (gfx1250) — compile-verified, matrix operand order checked by reading, not run on hardware
//
#include <hip/hip_runtime.h>

#define N_NODES 4096
#define N_EDGES 16384
#define N_GRAPH 16
#define HDIM    128
#define LN_EPS  1e-5f

typedef __bf16 v16bf  __attribute__((ext_vector_type(16)));
typedef __bf16 bf16x2 __attribute__((ext_vector_type(2)));
typedef float  v8f    __attribute__((ext_vector_type(8)));

// ---------- small helpers ----------
static __device__ __forceinline__ unsigned short f2bf(float f) {
    unsigned int u = __builtin_bit_cast(unsigned int, f);
    unsigned int lsb = (u >> 16) & 1u;
    u += 0x7fffu + lsb;                      // round-to-nearest-even
    return (unsigned short)(u >> 16);
}
static __device__ __forceinline__ float bf2f(unsigned short h) {
    unsigned int u = ((unsigned int)h) << 16;
    return __builtin_bit_cast(float, u);
}
// packed bf16x2 multiply -> v_pk_mul_bf16 (confirmed in asm)
static __device__ __forceinline__ unsigned int pkmul_bf16(unsigned int a, unsigned int b) {
    bf16x2 x = __builtin_bit_cast(bf16x2, a);
    bf16x2 y = __builtin_bit_cast(bf16x2, b);
    bf16x2 r = x * y;
    return __builtin_bit_cast(unsigned int, r);
}

// ---------- utility kernels ----------
__global__ void zero_f32_kernel(float* p, int n) {
    int i = blockIdx.x * 256 + threadIdx.x;
    if (i < n) p[i] = 0.f;
}
__global__ void f32_to_bf16_kernel(const float* __restrict__ in,
                                   unsigned short* __restrict__ out, int n) {
    int i = blockIdx.x * 256 + threadIdx.x;
    if (i < n) out[i] = f2bf(in[i]);
}
__global__ void degree_kernel(const int* __restrict__ dst, float* __restrict__ deg) {
    int e = blockIdx.x * 256 + threadIdx.x;
    if (e < N_EDGES) atomicAdd(&deg[dst[e]], 1.0f);
}
// xsum[n,c] = sum over edges with dst==n of feat[src[e],c]  (factored-out b2 path)
__global__ void xsum_kernel(const float* __restrict__ feat, int C_in,
                            const int* __restrict__ src, const int* __restrict__ dst,
                            float* __restrict__ xsum) {
    int id = blockIdx.x * 256 + threadIdx.x;
    if (id >= N_EDGES * C_in) return;
    int e = id / C_in, c = id - e * C_in;
    atomicAdd(&xsum[(size_t)dst[e] * C_in + c], feat[(size_t)src[e] * C_in + c]);
}

// ---------- edge MLP: h_e = relu(edge_attr @ w1 + b1), bf16 out ----------
__global__ void edge_mlp_kernel(const float* __restrict__ ea,
                                const float* __restrict__ w1,
                                const float* __restrict__ b1,
                                unsigned short* __restrict__ he) {
    int id = blockIdx.x * 256 + threadIdx.x;
    if (id >= N_EDGES * HDIM) return;
    int e = id >> 7, o = id & 127;
    float acc = b1[o];
#pragma unroll
    for (int c = 0; c < 4; ++c) acc += ea[e * 4 + c] * w1[c * HDIM + o];
    he[id] = f2bf(fmaxf(acc, 0.f));
}

// ---------- pack w2 into B-fragment layout ----------
__global__ void pack_w2_kernel(const float* __restrict__ w2,
                               unsigned int* __restrict__ out,
                               int C_in, int nkb) {
    int id = blockIdx.x * 256 + threadIdx.x;
    if (id >= nkb * 2048) return;
    int d    = id & 7;
    int lane = (id >> 3) & 31;
    int nt   = (id >> 8) & 7;
    int kb   = id >> 11;
    int half = lane >> 4, llo = lane & 15;
    int K0   = ((d < 4) ? (2 * d) : (2 * d + 8)) + 8 * half;
    int kap0 = kb * 32 + K0;
    int o    = nt * 16 + llo;
    int W    = C_in * HDIM;
    int i0 = kap0 >> 7,       k0 = kap0 & 127;
    int i1 = (kap0 + 1) >> 7, k1 = (kap0 + 1) & 127;
    unsigned int lo = f2bf(w2[k0 * W + i0 * HDIM + o]);
    unsigned int hi = f2bf(w2[k1 * W + i1 * HDIM + o]);
    out[id] = (hi << 16) | lo;
}

// ---------- fused message GEMM + scatter-add ----------
// Each wave owns TWO 16-edge tiles: every B fragment feeds 2 WMMAs (halves B
// traffic per MAC). K is split across blockIdx.y (outputs are atomic-summed).
template <int C_IN>
__global__ void __launch_bounds__(128) fused_msg_kernel(
    const unsigned short* __restrict__ nodebf,   // [N, C_IN] bf16
    const unsigned int*  __restrict__ heq,       // [E, 64] dwords (bf16 pairs)
    const unsigned int*  __restrict__ w2p,       // packed B fragments
    const int* __restrict__ src, const int* __restrict__ dst,
    float* __restrict__ s_acc) {
    constexpr int NKB = (C_IN * HDIM) / 32;
    constexpr int KBH = NKB / 2;                 // K-range per y-block
    __shared__ unsigned int xs_t[8 * 16 * 65];   // 8 tiles, stride 65: bank==row
    __shared__ unsigned int he_t[8 * 16 * 65];
    __shared__ int src_l[128];
    __shared__ int dst_l[128];

    const int tid = threadIdx.x;
    const int e0  = blockIdx.x * 128;
    src_l[tid] = src[e0 + tid];
    dst_l[tid] = dst[e0 + tid];
    __syncthreads();
    for (int idx = tid; idx < 8 * 16 * 64; idx += 128) {
        int t = idx >> 10, rem = idx & 1023;
        int mm = rem >> 6, c = rem & 63;
        he_t[t * 1040 + mm * 65 + c] = heq[(size_t)(e0 + t * 16 + mm) * 64 + c];
    }
    if (C_IN == HDIM) {
        const unsigned int* nq = (const unsigned int*)nodebf;
        for (int idx = tid; idx < 8 * 16 * 64; idx += 128) {
            int t = idx >> 10, rem = idx & 1023;
            int mm = rem >> 6, c = rem & 63;
            xs_t[t * 1040 + mm * 65 + c] = nq[(size_t)src_l[t * 16 + mm] * 64 + c];
        }
    } else {
        unsigned short* xsw = (unsigned short*)xs_t;
        for (int idx = tid; idx < 128 * C_IN; idx += 128) {
            int mg = idx / C_IN, c = idx - mg * C_IN;
            int t = mg >> 4, mm = mg & 15;
            xsw[t * 2080 + mm * 130 + c] = nodebf[(size_t)src_l[mg] * C_IN + c];
        }
    }
    __syncthreads();

    const int lane = tid & 31;
    const int wv   = tid >> 5;          // wave 0..3 -> edge tiles 2*wv, 2*wv+1
    const int half = lane >> 4;
    const int llo  = lane & 15;
    const int m    = llo;               // A-matrix row owned by this lane

    union { unsigned int u[8]; v16bf v; } A0, A1;
    union { unsigned int u[8]; v16bf v; } B;
    v8f acc0[8], acc1[8];
#pragma unroll
    for (int nt = 0; nt < 8; ++nt) { acc0[nt] = {}; acc1[nt] = {}; }

    const unsigned short* xs0 = ((const unsigned short*)xs_t) + (2 * wv) * 2080;
    const unsigned short* xs1 = xs0 + 2080;
    const unsigned int*   h0  = he_t + (2 * wv) * 1040 + m * 65;
    const unsigned int*   h1  = h0 + 1040;

    const int kb_lo = blockIdx.y * KBH;
    for (int kb = kb_lo; kb < kb_lo + KBH; ++kb) {
        const int kap0  = kb * 32;
        const int i_idx = kap0 >> 7;          // constant x-index for this K-block
        const int kpb   = (kap0 & 127) >> 1;  // k' base in dword pairs
        unsigned short xv0 = xs0[m * 130 + i_idx];
        unsigned short xv1 = xs1[m * 130 + i_idx];
        unsigned int xw0 = ((unsigned int)xv0 << 16) | xv0;
        unsigned int xw1 = ((unsigned int)xv1 << 16) | xv1;
#pragma unroll
        for (int d = 0; d < 8; ++d) {
            int K0 = ((d < 4) ? (2 * d) : (2 * d + 8)) + 8 * half;
            unsigned int hk0 = h0[kpb + (K0 >> 1)];
            unsigned int hk1 = h1[kpb + (K0 >> 1)];
            A0.u[d] = pkmul_bf16(xw0, hk0);
            A1.u[d] = pkmul_bf16(xw1, hk1);
        }
        if (kb + 1 < kb_lo + KBH)   // stream next K-block's B fragments in
            __builtin_prefetch(w2p + (((size_t)(kb + 1) * 8 * 32 + lane) * 8), 0, 3);
#pragma unroll
        for (int nt = 0; nt < 8; ++nt) {
            const unsigned int* bp = w2p + (((size_t)(kb * 8 + nt) * 32 + lane) * 8);
#pragma unroll
            for (int d = 0; d < 8; ++d) B.u[d] = bp[d];
            acc0[nt] = __builtin_amdgcn_wmma_f32_16x16x32_bf16(
                false, A0.v, false, B.v, (short)0, acc0[nt], false, false);
            acc1[nt] = __builtin_amdgcn_wmma_f32_16x16x32_bf16(
                false, A1.v, false, B.v, (short)0, acc1[nt], false, false);
        }
    }

#pragma unroll
    for (int nt = 0; nt < 8; ++nt) {
        const int o = nt * 16 + llo;
#pragma unroll
        for (int r = 0; r < 8; ++r) {
            int mm = r + 8 * half;
            atomicAdd(&s_acc[(size_t)dst_l[(2 * wv) * 16 + mm] * HDIM + o], acc0[nt][r]);
            atomicAdd(&s_acc[(size_t)dst_l[(2 * wv + 1) * 16 + mm] * HDIM + o], acc1[nt][r]);
        }
    }
}

// ---------- node update: mean-agg (+xsum@b2) + root matmul + bias + LN + act ----------
__global__ void __launch_bounds__(128) node_update_kernel(
    const float* __restrict__ s_acc, const float* __restrict__ deg,
    const float* __restrict__ xsum, const float* __restrict__ b2,
    const float* __restrict__ in_feat, int C_in,
    const float* __restrict__ root, const float* __restrict__ bias,
    const float* __restrict__ g, const float* __restrict__ b,
    float* __restrict__ out_f, unsigned short* __restrict__ out_bf, int act) {
    __shared__ float red[128];
    int n = blockIdx.x, o = threadIdx.x;
    float acc_s = s_acc[(size_t)n * HDIM + o];
    const float* xsr = xsum + (size_t)n * C_in;
    for (int c = 0; c < C_in; ++c) acc_s += xsr[c] * b2[c * HDIM + o];
    float a = acc_s / fmaxf(deg[n], 1.0f);
    const float* xr = in_feat + (size_t)n * C_in;
    for (int c = 0; c < C_in; ++c) a += xr[c] * root[c * HDIM + o];
    a += bias[o];
    red[o] = a; __syncthreads();
    for (int st = 64; st > 0; st >>= 1) { if (o < st) red[o] += red[o + st]; __syncthreads(); }
    float mean = red[0] * (1.f / 128.f); __syncthreads();
    float dm = a - mean;
    red[o] = dm * dm; __syncthreads();
    for (int st = 64; st > 0; st >>= 1) { if (o < st) red[o] += red[o + st]; __syncthreads(); }
    float var = red[0] * (1.f / 128.f);
    float y = dm * rsqrtf(var + LN_EPS) * g[o] + b[o];
    if (act == 0)      y = fmaxf(y, 0.f);
    else if (act == 1) y = (y > 0.f) ? y : (expf(y) - 1.f);
    else               y = (y >= 0.f) ? y : 0.01f * y;
    out_f[(size_t)n * HDIM + o] = y;
    out_bf[(size_t)n * HDIM + o] = f2bf(y);
}

// ---------- per-graph mean/max pooling ----------
__global__ void __launch_bounds__(128) pool_kernel(
    const float* __restrict__ h, const int* __restrict__ batch,
    float* __restrict__ gmean, float* __restrict__ gmax) {
    int g = blockIdx.x, o = threadIdx.x;
    float s = 0.f, mx = -3.402823466e38f, cnt = 0.f;
    for (int n = 0; n < N_NODES; ++n) {
        if (batch[n] == g) {
            float v = h[(size_t)n * HDIM + o];
            s += v; mx = fmaxf(mx, v); cnt += 1.f;
        }
    }
    gmean[g * HDIM + o] = s / fmaxf(cnt, 1.f);
    gmax[g * HDIM + o]  = (cnt > 0.f) ? mx : 0.f;
}

// ---------- final linear head ----------
__global__ void head_kernel(const float* __restrict__ gmean, const float* __restrict__ gmax,
                            const float* __restrict__ cont, const float* __restrict__ ls,
                            const float* __restrict__ w, const float* __restrict__ bb,
                            float* __restrict__ out) {
    int t = threadIdx.x;
    if (t >= 2 * N_GRAPH) return;
    int g = t >> 1, j = t & 1;
    float acc = bb[j];
    for (int c = 0; c < 2 * HDIM + 5; ++c) {
        float v;
        if (c < HDIM)            v = gmean[g * HDIM + c];
        else if (c < 2 * HDIM)   v = gmax[g * HDIM + (c - HDIM)];
        else if (c < 2 * HDIM+4) v = cont[g * 4 + (c - 2 * HDIM)];
        else                     v = ls[g];
        acc += v * w[c * 2 + j];
    }
    out[g * 2 + j] = acc;
}

extern "C" void kernel_launch(void* const* d_in, const int* in_sizes, int n_in,
                              void* d_out, int out_size, void* d_ws, size_t ws_size,
                              hipStream_t stream) {
    (void)in_sizes; (void)n_in; (void)out_size; (void)ws_size;
    const float* x      = (const float*)d_in[0];
    const int*   ei     = (const int*)  d_in[1];
    const float* ea     = (const float*)d_in[2];
    const int*   batch  = (const int*)  d_in[3];
    const float* cont   = (const float*)d_in[4];
    const float* lscale = (const float*)d_in[5];
    const float* w1_[3]   = {(const float*)d_in[6],  (const float*)d_in[12], (const float*)d_in[18]};
    const float* b1_[3]   = {(const float*)d_in[7],  (const float*)d_in[13], (const float*)d_in[19]};
    const float* w2_[3]   = {(const float*)d_in[8],  (const float*)d_in[14], (const float*)d_in[20]};
    const float* b2_[3]   = {(const float*)d_in[9],  (const float*)d_in[15], (const float*)d_in[21]};
    const float* root_[3] = {(const float*)d_in[10], (const float*)d_in[16], (const float*)d_in[22]};
    const float* bias_[3] = {(const float*)d_in[11], (const float*)d_in[17], (const float*)d_in[23]};
    const float* lng_[3]  = {(const float*)d_in[24], (const float*)d_in[26], (const float*)d_in[28]};
    const float* lnb_[3]  = {(const float*)d_in[25], (const float*)d_in[27], (const float*)d_in[29]};
    const float* lin_w  = (const float*)d_in[30];
    const float* lin_b  = (const float*)d_in[31];
    const int* src = ei;
    const int* dst = ei + N_EDGES;

    // workspace layout
    char* ws = (char*)d_ws;
    size_t off = 0;
    auto take = [&](size_t bytes) { char* p = ws + off; off += (bytes + 255) & ~(size_t)255; return p; };
    unsigned short* nodebf = (unsigned short*)take((size_t)N_NODES * HDIM * 2);
    unsigned short* he     = (unsigned short*)take((size_t)N_EDGES * HDIM * 2);
    unsigned int*   w2p    = (unsigned int*)  take((size_t)512 * 2048 * 4);
    float*          s_acc  = (float*)         take((size_t)N_NODES * HDIM * 4);
    float*          xsum   = (float*)         take((size_t)N_NODES * HDIM * 4);
    float*          h_a    = (float*)         take((size_t)N_NODES * HDIM * 4);
    float*          h_b    = (float*)         take((size_t)N_NODES * HDIM * 4);
    float*          deg    = (float*)         take((size_t)N_NODES * 4);
    float*          gmean  = (float*)         take((size_t)N_GRAPH * HDIM * 4);
    float*          gmax   = (float*)         take((size_t)N_GRAPH * HDIM * 4);

    // degrees (shared by all layers)
    zero_f32_kernel<<<(N_NODES + 255) / 256, 256, 0, stream>>>(deg, N_NODES);
    degree_kernel<<<(N_EDGES + 255) / 256, 256, 0, stream>>>(dst, deg);
    // x -> bf16 (stride 3 rows)
    f32_to_bf16_kernel<<<(N_NODES * 3 + 255) / 256, 256, 0, stream>>>(x, nodebf, N_NODES * 3);

    const int   cin_[3]  = {3, HDIM, HDIM};
    const float* inf_[3] = {x, h_a, h_b};
    float*       outf_[3]= {h_a, h_b, h_a};

    for (int L = 0; L < 3; ++L) {
        int C_in = cin_[L];
        int nkb  = (C_in * HDIM) / 32;
        edge_mlp_kernel<<<(N_EDGES * HDIM) / 256, 256, 0, stream>>>(ea, w1_[L], b1_[L], he);
        pack_w2_kernel<<<(nkb * 2048 + 255) / 256, 256, 0, stream>>>(w2_[L], w2p, C_in, nkb);
        zero_f32_kernel<<<(N_NODES * HDIM) / 256, 256, 0, stream>>>(s_acc, N_NODES * HDIM);
        zero_f32_kernel<<<(N_NODES * C_in + 255) / 256, 256, 0, stream>>>(xsum, N_NODES * C_in);
        xsum_kernel<<<(N_EDGES * C_in + 255) / 256, 256, 0, stream>>>(inf_[L], C_in, src, dst, xsum);
        dim3 mg(N_EDGES / 128, 2);   // K split across y (atomic-summed outputs)
        if (L == 0)
            fused_msg_kernel<3><<<mg, 128, 0, stream>>>(
                nodebf, (const unsigned int*)he, w2p, src, dst, s_acc);
        else
            fused_msg_kernel<HDIM><<<mg, 128, 0, stream>>>(
                nodebf, (const unsigned int*)he, w2p, src, dst, s_acc);
        node_update_kernel<<<N_NODES, 128, 0, stream>>>(
            s_acc, deg, xsum, b2_[L], inf_[L], C_in, root_[L], bias_[L], lng_[L], lnb_[L],
            outf_[L], nodebf, L);
    }

    pool_kernel<<<N_GRAPH, 128, 0, stream>>>(h_a, batch, gmean, gmax);
    head_kernel<<<1, 32, 0, stream>>>(gmean, gmax, cont, lscale, lin_w, lin_b, (float*)d_out);
}
